// FFTMLP_77386720739464
// MI455X (gfx1250) — matrix operand order repaired and checked by reading, not compiled, not run on hardware
//
#include <hip/hip_runtime.h>

// ---------------------------------------------------------------------------
// Out(8192x4096) = X(8192x2049) @ filters_real(4096x2049)^T   (fp32 in/out)
// bf16 WMMA (v_wmma_f32_16x16x32_bf16) with f32 accumulate.
// Double-buffered LDS; workgroup barrier via inline asm that waits ONLY on
// DScnt so tile-t global loads stay in flight across the barrier and drain
// under the WMMAs (CDNA5 split counters: LOADcnt vs DScnt independent).
// ---------------------------------------------------------------------------

typedef __attribute__((ext_vector_type(16))) __bf16 v16bf;
typedef __attribute__((ext_vector_type(8)))  __bf16 v8bf;
typedef __attribute__((ext_vector_type(2)))  __bf16 v2bf;
typedef __attribute__((ext_vector_type(8)))  float  v8f;

#define M_DIM 8192
#define N_DIM 4096
#define K_DIM 2049
#define KT    32           // WMMA K per tile
#define BM    128          // workgroup M tile
#define BN    128          // workgroup N tile
#define LDT   40           // padded LDS row stride (bf16): 80B, 16B-aligned, conflict-free
#define NFULL 64           // full K tiles (k = 0..2047); tail tile has 1 valid column

// Workgroup barrier that drains only the DS counter (LDS store visibility +
// our own ds_loads complete) but leaves global loads in flight.
#define WG_BARRIER_DS_ONLY()                                  \
    asm volatile("s_wait_dscnt 0x0\n\t"                       \
                 "s_barrier_signal -1\n\t"                    \
                 "s_barrier_wait -1" ::: "memory")

__global__ __launch_bounds__(256) void fftmlp_gemm_bf16wmma(
    const float* __restrict__ X,   // (M, K)
    const float* __restrict__ W,   // (N, K)
    float* __restrict__ Out)       // (M, N)
{
    __shared__ __bf16 sA[2][BM * LDT];
    __shared__ __bf16 sB[2][BN * LDT];

    const int tid  = threadIdx.x;
    const int lane = tid & 31;
    const int wave = tid >> 5;

    const int mWG = blockIdx.y * BM;
    const int nWG = blockIdx.x * BN;

    // 8 waves: 4 along M (32 rows) x 2 along N (64 cols)
    const int wr = wave >> 1;
    const int wc = wave & 1;

    // loader: each thread owns 2 consecutive k-columns of 8 rows (per matrix)
    const int kp = tid & 15;        // k-pair index: k = 2*kp, 2*kp+1
    const int r0 = tid >> 4;        // 0..15; rows r = r0 + 16*i

    // WMMA fragment lane decomposition (ISA 7.12.2, wave32)
    const int half = lane >> 4;     // 0 | 1
    const int fl   = lane & 15;

    v8f acc[2][4];
    const v8f vzero = {};
#pragma unroll
    for (int i = 0; i < 2; ++i)
#pragma unroll
        for (int j = 0; j < 4; ++j) acc[i][j] = vzero;

    // register staging for the global->LDS pipeline
    float fa[8][2], fb[8][2];

    // ---- batched unguarded fetch of one K-tile into registers
    auto fetch = [&](int k0) {
        const int kg = k0 + 2 * kp;
#pragma unroll
        for (int i = 0; i < 8; ++i) {
            const int r = r0 + 16 * i;
            const float* xa = &X[(size_t)(mWG + r) * K_DIM + kg];
            const float* xw = &W[(size_t)(nWG + r) * K_DIM + kg];
            fa[i][0] = xa[0]; fa[i][1] = xa[1];
            fb[i][0] = xw[0]; fb[i][1] = xw[1];
        }
    };

    // ---- guarded fetch (only for the final partial tile, k0 = 2048)
    auto fetch_tail = [&](int k0) {
        const int kg = k0 + 2 * kp;
        const bool ok0 = (kg     < K_DIM);
        const bool ok1 = (kg + 1 < K_DIM);
#pragma unroll
        for (int i = 0; i < 8; ++i) {
            const int r = r0 + 16 * i;
            const float* xa = &X[(size_t)(mWG + r) * K_DIM + kg];
            const float* xw = &W[(size_t)(nWG + r) * K_DIM + kg];
            fa[i][0] = ok0 ? xa[0] : 0.0f;  fa[i][1] = ok1 ? xa[1] : 0.0f;
            fb[i][0] = ok0 ? xw[0] : 0.0f;  fb[i][1] = ok1 ? xw[1] : 0.0f;
        }
    };

    // ---- convert + packed bf16x2 stores into LDS buffer `buf`
    auto stage = [&](int buf) {
#pragma unroll
        for (int i = 0; i < 8; ++i) {
            const int r = r0 + 16 * i;
            v2bf pa, pb;
            pa.x = (__bf16)fa[i][0]; pa.y = (__bf16)fa[i][1];
            pb.x = (__bf16)fb[i][0]; pb.y = (__bf16)fb[i][1];
            *(v2bf*)&sA[buf][r * LDT + 2 * kp] = pa;   // 4B-aligned ds_store_b32
            *(v2bf*)&sB[buf][r * LDT + 2 * kp] = pb;
        }
    };

    // ---- 8 WMMAs from LDS buffer `buf`
    auto compute = [&](int buf) {
        v16bf afr[2];
#pragma unroll
        for (int mt = 0; mt < 2; ++mt) {
            // A 16x32 bf16: lane half selects K {0..7,16..23} vs {8..15,24..31}
            const __bf16* src = &sA[buf][(wr * 32 + mt * 16 + fl) * LDT + half * 8];
            v8bf lo = *(const v8bf*)(src);
            v8bf hi = *(const v8bf*)(src + 16);
            afr[mt] = __builtin_shufflevector(lo, hi,
                        0, 1, 2, 3, 4, 5, 6, 7, 8, 9, 10, 11, 12, 13, 14, 15);
        }
        v16bf bfr[4];
#pragma unroll
        for (int nt = 0; nt < 4; ++nt) {
            // B 32x16 bf16: lane fl = column n, half selects K {0..7,16..23} vs {8..15,24..31}
            const __bf16* src = &sB[buf][(wc * 64 + nt * 16 + fl) * LDT + half * 8];
            v8bf lo = *(const v8bf*)(src);
            v8bf hi = *(const v8bf*)(src + 16);
            bfr[nt] = __builtin_shufflevector(lo, hi,
                        0, 1, 2, 3, 4, 5, 6, 7, 8, 9, 10, 11, 12, 13, 14, 15);
        }
#pragma unroll
        for (int mt = 0; mt < 2; ++mt)
#pragma unroll
            for (int nt = 0; nt < 4; ++nt)
                acc[mt][nt] = __builtin_amdgcn_wmma_f32_16x16x32_bf16(
                    false, afr[mt], false, bfr[nt],
                    (short)0, acc[mt][nt], false, false);
    };

    // =================== pipelined main loop ===================
    // tile t lives in LDS buffer (t & 1); tiles 0..63 full, tile 64 = K tail
    fetch(0);
    stage(0);

    for (int t = 1; t < NFULL; ++t) {
        fetch(t * KT);            // issue tile-t global loads (stay in flight
        WG_BARRIER_DS_ONLY();     //  across the barrier: only DScnt drained)
        compute((t - 1) & 1);     // WMMAs overlap outstanding global loads
        stage(t & 1);             // loadcnt waits land here, after the WMMAs
    }

    // tail tile (t = 64, k0 = 2048: one valid column, rest zero-padded)
    fetch_tail(NFULL * KT);
    WG_BARRIER_DS_ONLY();
    compute((NFULL - 1) & 1);     // tile 63 (buffer 1)
    stage(NFULL & 1);             // tile 64 -> buffer 0
    WG_BARRIER_DS_ONLY();
    compute(NFULL & 1);           // tile 64

    // =================== store C ===================
    // C/D layout: VGPR r -> row M = r + 8*half, lane fl -> column N
#pragma unroll
    for (int mt = 0; mt < 2; ++mt) {
#pragma unroll
        for (int nt = 0; nt < 4; ++nt) {
            const int n = nWG + wc * 64 + nt * 16 + fl;
#pragma unroll
            for (int r = 0; r < 8; ++r) {
                const int m = mWG + wr * 32 + mt * 16 + r + 8 * half;
                Out[(size_t)m * N_DIM + n] = acc[mt][nt][r];
            }
        }
    }
}

extern "C" void kernel_launch(void* const* d_in, const int* in_sizes, int n_in,
                              void* d_out, int out_size, void* d_ws, size_t ws_size,
                              hipStream_t stream) {
    const float* x  = (const float*)d_in[0];  // (8192, 2049)
    const float* wr = (const float*)d_in[1];  // (4096, 2049) filters_real
    // d_in[2] = filters_imag: never contributes (x is real)
    float* out = (float*)d_out;               // (8192, 4096)

    dim3 grid(N_DIM / BN, M_DIM / BM);        // (32, 64)
    dim3 block(256);
    fftmlp_gemm_bf16wmma<<<grid, block, 0, stream>>>(x, wr, out);
}
